// GHGEAT_wo_88072599372334
// MI455X (gfx1250) — matrix operand order repaired and checked by reading, not compiled, not run on hardware
//
#include <hip/hip_runtime.h>
#include <hip/hip_fp16.h>

typedef __attribute__((ext_vector_type(16))) _Float16 v16h;
typedef __attribute__((ext_vector_type(8)))  _Float16 v8h;
typedef __attribute__((ext_vector_type(8)))  float    v8f;
typedef _Float16 h16;

#define NNODES 102400
#define NEDGES 204800
#define NB     2048
#define NPG    50
#define EPG    100
#define EPSV   1e-5f

__device__ __forceinline__ float warp_sum(float v){
  #pragma unroll
  for (int o = 16; o; o >>= 1) v += __shfl_xor(v, o, 32);
  return v;
}
__device__ __forceinline__ float warp_max(float v){
  #pragma unroll
  for (int o = 16; o; o >>= 1) v = fmaxf(v, __shfl_xor(v, o, 32));
  return v;
}

__device__ __forceinline__ v16h ld_frag(const h16* p){
  v8h lo = *(const v8h*)p;
  v8h hi = *(const v8h*)(p + 8);
  return __builtin_shufflevector(lo, hi, 0,1,2,3,4,5,6,7,8,9,10,11,12,13,14,15);
}

// =============================================================================
// WMMA GEMM: C[M,N] = A[M,Kpad](f16,row-major) @ W(f16, pre-swizzled) + bias
// Weight swizzle layout: [N/16][Kpad/32][lane 0..31][16 halves]: lane l of
// k-block kb holds B[kb*32 + (l>>4)*16 + i][tn*16 + (l&15)], i=0..15 — the
// B fragment in contiguous k order (A uses the same), loaded as 2 contiguous b128.
// Each wave computes a 32 x (16*NT) C block: 2 M-tiles x NT N-tiles, sharing
// A/B fragments across tiles -> 1.5 b128 loads per WMMA (NT=4), and 2*NT
// independent accumulators to cover the F16 WMMA RAW hazard window.
// =============================================================================
template<int NT>
__global__ void k_gemm(const h16* __restrict__ A, int lda,
                       const h16* __restrict__ Wsw, int N, int kbCnt,
                       const float* __restrict__ bias,
                       float* __restrict__ Cf, h16* __restrict__ Ch, int ldc,
                       int M, int relu)
{
  int wid  = (blockIdx.x * blockDim.x + threadIdx.x) >> 5;
  int lane = threadIdx.x & 31;
  int tilesNg = N / (16 * NT);
  long long tiles = (long long)(M >> 5) * tilesNg;     // 32 rows per wave
  if (wid >= tiles) return;
  int tm  = wid / tilesNg;
  int tng = wid - tm * tilesNg;
  int hi  = lane >> 4;
  int l15 = lane & 15;

  const h16* ap0 = A + (size_t)(tm * 32 + l15) * lda + hi * 16;
  const h16* ap1 = ap0 + (size_t)16 * lda;
  const h16* wb  = Wsw + ((size_t)tng * NT * kbCnt) * 512 + lane * 16;

  v8f z = {};
  v8f acc[2][NT];
  #pragma unroll
  for (int t = 0; t < NT; t++) { acc[0][t] = z; acc[1][t] = z; }

  for (int kb = 0; kb < kbCnt; kb++) {
    v16h a0 = ld_frag(ap0 + kb * 32);
    v16h a1 = ld_frag(ap1 + kb * 32);
    #pragma unroll
    for (int t = 0; t < NT; t++) {
      v16h b = ld_frag(wb + ((size_t)t * kbCnt + kb) * 512);
      acc[0][t] = __builtin_amdgcn_wmma_f32_16x16x32_f16(false, a0, false, b,
                                                         (short)0, acc[0][t], false, false);
      acc[1][t] = __builtin_amdgcn_wmma_f32_16x16x32_f16(false, a1, false, b,
                                                         (short)0, acc[1][t], false, false);
    }
  }

  #pragma unroll
  for (int m2 = 0; m2 < 2; m2++) {
    int crow = tm * 32 + m2 * 16 + hi * 8;
    #pragma unroll
    for (int t = 0; t < NT; t++) {
      int ncol = (tng * NT + t) * 16 + l15;
      float bv = bias ? bias[ncol] : 0.0f;
      #pragma unroll
      for (int r = 0; r < 8; r++) {
        float v = acc[m2][t][r] + bv;
        if (relu) v = fmaxf(v, 0.0f);
        size_t o = (size_t)(crow + r) * ldc + ncol;
        if (Cf) Cf[o] = v;
        if (Ch) Ch[o] = (h16)v;
      }
    }
  }
}

// --------- weight convert f32 -> f16 B-fragment-swizzled [N/16][Kpad/32][32][16]
// mode 0: src row-major [K0,N0]; mode 1: src is [N0,K0] (use its transpose);
// mode 2: NNConv W2' reorder: W[k][n] = w2[(n>>7)*16384 + k*128 + (n&127)]
__global__ void k_wswz(const float* __restrict__ src, h16* __restrict__ dst,
                       int K0, int N0, int Kpad, int N, int mode)
{
  int idx = blockIdx.x * blockDim.x + threadIdx.x;
  int kbCnt = Kpad >> 5;
  long long total = (long long)(N >> 4) * kbCnt * 512;
  if (idx >= total) return;
  int i    = idx & 15;
  int lane = (idx >> 4) & 31;
  int rem  = idx >> 9;
  int kb   = rem % kbCnt;
  int tn   = rem / kbCnt;
  int k = kb * 32 + (lane >> 4) * 16 + i;
  int n = tn * 16 + (lane & 15);
  float v = 0.f;
  if (k < K0 && n < N0) {
    if (mode == 0)      v = src[(size_t)k * N0 + n];
    else if (mode == 1) v = src[(size_t)n * K0 + k];
    else                v = src[(size_t)(n >> 7) * 16384 + (size_t)k * 128 + (n & 127)];
  }
  dst[idx] = (h16)v;
}

// --------- edge MLP input gather: [x[row] | x[col] | ea | u[batch[row]] | pad] --
__global__ void k_edge_in(const float* __restrict__ x, const float* __restrict__ ea, int eaDim,
                          const float* __restrict__ u, int uDim,
                          const int* __restrict__ row, const int* __restrict__ col,
                          const int* __restrict__ batch, h16* __restrict__ out, int Kpad)
{
  long long idx = blockIdx.x * (long long)blockDim.x + threadIdx.x;
  if (idx >= (long long)NEDGES * Kpad) return;
  int e = (int)(idx / Kpad), j = (int)(idx - (long long)e * Kpad);
  float v = 0.f;
  if (j < 64)                v = x[(size_t)row[e] * 64 + j];
  else if (j < 128)          v = x[(size_t)col[e] * 64 + (j - 64)];
  else if (j < 128 + eaDim)  v = ea[(size_t)e * eaDim + (j - 128)];
  else { int jj = j - 128 - eaDim; if (jj < uDim) v = u[(size_t)batch[row[e]] * uDim + jj]; }
  out[idx] = (h16)v;
}

__global__ void k_zero(float* p, long long n){
  long long i = blockIdx.x * (long long)blockDim.x + threadIdx.x;
  if (i < n) p[i] = 0.f;
}

__global__ void k_scatter(const float* __restrict__ ea, const int* __restrict__ col,
                          float* __restrict__ agg)
{
  int idx = blockIdx.x * blockDim.x + threadIdx.x;
  if (idx >= NEDGES * 64) return;
  int e = idx >> 6, f = idx & 63;
  atomicAdd(&agg[(size_t)col[e] * 64 + f], ea[idx]);
}

// --------- node update input: concat + layernorm -> f32 + f16 (zero-padded) ----
__global__ void k_node_ln(const float* __restrict__ x, const float* __restrict__ agg,
                          const float* __restrict__ u, int uDim, const int* __restrict__ batch,
                          const float* __restrict__ g, const float* __restrict__ b,
                          float* __restrict__ outF, h16* __restrict__ outH, int d1, int LD)
{
  int node = (blockIdx.x * blockDim.x + threadIdx.x) >> 5;
  int lane = threadIdx.x & 31;
  if (node >= NNODES) return;
  int nt = LD >> 5;
  float vals[6];
  int bg = batch[node];
  for (int t = 0; t < nt; t++) {
    int j = lane + (t << 5);
    float v = 0.f;
    if (j < 64)       v = x[(size_t)node * 64 + j];
    else if (j < 128) v = agg[(size_t)node * 64 + (j - 64)];
    else if (j < d1)  v = u[(size_t)bg * uDim + (j - 128)];
    vals[t] = v;
  }
  float s = 0.f;
  for (int t = 0; t < nt; t++) s += vals[t];
  float mean = warp_sum(s) / (float)d1;
  float sq = 0.f;
  for (int t = 0; t < nt; t++) {
    int j = lane + (t << 5);
    float d = (j < d1) ? (vals[t] - mean) : 0.f;
    sq += d * d;
  }
  float inv = rsqrtf(warp_sum(sq) / (float)d1 + EPSV);
  for (int t = 0; t < nt; t++) {
    int j = lane + (t << 5);
    float o = (j < d1) ? (vals[t] - mean) * inv * g[j] + b[j] : 0.f;
    outF[(size_t)node * LD + j] = o;
    outH[(size_t)node * LD + j] = (h16)o;
  }
}

__global__ void k_softmax128(const float* __restrict__ logit, h16* __restrict__ prob)
{
  int row  = (blockIdx.x * blockDim.x + threadIdx.x) >> 5;
  int lane = threadIdx.x & 31;
  if (row >= NNODES) return;
  const float* lp = logit + (size_t)row * 128;
  float v[4];
  float m = -1e30f;
  #pragma unroll
  for (int t = 0; t < 4; t++) { v[t] = lp[lane + 32 * t]; m = fmaxf(m, v[t]); }
  m = warp_max(m);
  float s = 0.f;
  #pragma unroll
  for (int t = 0; t < 4; t++) { v[t] = __expf(v[t] - m); s += v[t]; }
  float inv = 1.0f / warp_sum(s);
  h16* pp = prob + (size_t)row * 128;
  #pragma unroll
  for (int t = 0; t < 4; t++) pp[lane + 32 * t] = (h16)(v[t] * inv);
}

__global__ void k_mix(const float* __restrict__ attn, const float* __restrict__ outF,
                      h16* __restrict__ outH, long long n)
{
  long long i = blockIdx.x * (long long)blockDim.x + threadIdx.x;
  if (i < n) outH[i] = (h16)(0.5f * (attn[i] + outF[i]));   // ALPHA = 0.5
}

__global__ void k_seg_mean(const float* __restrict__ x, float* __restrict__ out, int rps)
{
  int g = blockIdx.x, f = threadIdx.x;
  const float* p = x + (size_t)g * rps * 64 + f;
  float s = 0.f;
  for (int i = 0; i < rps; i++) s += p[(size_t)i * 64];
  out[(size_t)g * 64 + f] = s / (float)rps;
}

__global__ void k_glob_in(const float* __restrict__ u, int uDim, const float* __restrict__ nagg,
                          const float* __restrict__ eagg, h16* __restrict__ out, int Kpad)
{
  int idx = blockIdx.x * blockDim.x + threadIdx.x;
  if (idx >= NB * Kpad) return;
  int g = idx / Kpad, j = idx - g * Kpad;
  float v = 0.f;
  if (j < uDim)             v = u[(size_t)g * uDim + j];
  else if (j < uDim + 64)   v = nagg[(size_t)g * 64 + (j - uDim)];
  else if (j < uDim + 128)  v = eagg[(size_t)g * 64 + (j - uDim - 64)];
  out[idx] = (h16)v;
}

__global__ void k_gnorm(float* __restrict__ x, const float* __restrict__ w,
                        const float* __restrict__ b, const float* __restrict__ ms)
{
  int g = blockIdx.x, f = threadIdx.x;
  float* p = x + (size_t)g * NPG * 64 + f;
  float s = 0.f;
  for (int i = 0; i < NPG; i++) s += p[(size_t)i * 64];
  float m = s / (float)NPG;
  float msf = ms[f];
  float var = 0.f;
  for (int i = 0; i < NPG; i++) { float o = p[(size_t)i * 64] - msf * m; var += o * o; }
  var /= (float)NPG;
  float inv = rsqrtf(var + EPSV);
  float wf = w[f], bf = b[f];
  for (int i = 0; i < NPG; i++) {
    float o = p[(size_t)i * 64] - msf * m;
    p[(size_t)i * 64] = o * inv * wf + bf;
  }
}

// ------------------------- system graph kernels -------------------------------
__global__ void k_nodefeat(const float* __restrict__ xg1, const float* __restrict__ u1,
                           const float* __restrict__ xg2, const float* __restrict__ u2,
                           h16* __restrict__ nf)
{
  int idx = blockIdx.x * blockDim.x + threadIdx.x;
  if (idx >= 2 * NB * 128) return;
  int n = idx >> 7, j = idx & 127;
  const float* xs = (n < NB) ? xg1 : xg2;
  const float* us = (n < NB) ? u1  : u2;
  int i = (n < NB) ? n : n - NB;
  float v = (j < 64) ? xs[(size_t)i * 64 + j] : us[(size_t)i * 64 + (j - 64)];
  nf[idx] = (h16)v;
}

__global__ void k_ewh(const float* __restrict__ hb1, const float* __restrict__ hb2,
                      const float* __restrict__ w1, const float* __restrict__ b1,
                      float* __restrict__ ewh)
{
  int idx = blockIdx.x * blockDim.x + threadIdx.x;
  if (idx >= 4 * NB * 32) return;
  int e = idx >> 5, k = idx & 31;
  float hb = (e < 3 * NB) ? hb1[e % NB] : hb2[e - 3 * NB];
  ewh[idx] = fmaxf(hb * w1[k] + b1[k], 0.f);
}

// conv += msg(edge1)+msg(edge2); x_t = relu(conv) -> f16.  Each node has exactly
// one cross edge + one self loop by construction, so no atomics.
__global__ void k_convmsg(const float* __restrict__ conv, const float* __restrict__ Q,
                          const float* __restrict__ P, const float* __restrict__ ewh,
                          h16* __restrict__ xt16)
{
  int n = blockIdx.x, o = threadIdx.x;
  int e1, r1, e2, r2;
  if (n < NB) { e1 = NB + n; r1 = NB + n; e2 = 2 * NB + n;        r2 = n; }
  else        { int i = n - NB; e1 = i; r1 = i; e2 = 3 * NB + i;  r2 = n; }
  __shared__ float w1s[32], w2s[32];
  if (threadIdx.x < 32)      w1s[threadIdx.x]      = ewh[(size_t)e1 * 32 + threadIdx.x];
  else if (threadIdx.x < 64) w2s[threadIdx.x - 32] = ewh[(size_t)e2 * 32 + (threadIdx.x - 32)];
  __syncthreads();
  float acc = conv[(size_t)n * 128 + o] + Q[(size_t)r1 * 128 + o] + Q[(size_t)r2 * 128 + o];
  const float* P1 = P + (size_t)r1 * 4096 + o;
  const float* P2 = P + (size_t)r2 * 4096 + o;
  #pragma unroll 8
  for (int k = 0; k < 32; k++) acc += w1s[k] * P1[(size_t)k * 128] + w2s[k] * P2[(size_t)k * 128];
  xt16[(size_t)n * 128 + o] = (h16)fmaxf(acc, 0.f);
}

__global__ void k_gru(const float* __restrict__ gi, const float* __restrict__ gh,
                      const float* __restrict__ h, h16* __restrict__ xg16)
{
  int idx = blockIdx.x * blockDim.x + threadIdx.x;
  if (idx >= 2 * NB * 128) return;
  int n = idx >> 7, o = idx & 127;
  const float* gin = gi + (size_t)n * 384;
  const float* ghn = gh + (size_t)n * 384;
  float r  = 1.f / (1.f + __expf(-(gin[o] + ghn[o])));
  float z  = 1.f / (1.f + __expf(-(gin[128 + o] + ghn[128 + o])));
  float nc = tanhf(gin[256 + o] + r * ghn[256 + o]);
  float hn = (1.f - z) * nc + z * h[idx];
  if (n < NB) xg16[(size_t)n * 256 + o]              = (h16)hn;
  else        xg16[(size_t)(n - NB) * 256 + 128 + o] = (h16)hn;
}

__global__ void k_head(const float* __restrict__ A2a, const float* __restrict__ A2b,
                       const float* __restrict__ w3a, const float* __restrict__ b3a,
                       const float* __restrict__ w3b, const float* __restrict__ b3b,
                       const float* __restrict__ T, float* __restrict__ out)
{
  int i = blockIdx.x * blockDim.x + threadIdx.x;
  if (i >= NB) return;
  float a = b3a[0], b = b3b[0];
  for (int j = 0; j < 64; j++) {
    a += A2a[(size_t)i * 64 + j] * w3a[j];
    b += A2b[(size_t)i * 64 + j] * w3b[j];
  }
  out[i] = a + b / (T[i] + 273.15f);
}

// =============================== host side ====================================
static void gemm(hipStream_t s, const h16* A, int lda, const h16* Wsw, int N,
                 const float* bias, float* Cf, h16* Ch, int ldc, int M, int Kpad, int relu)
{
  int kb = Kpad / 32;
  if (N % 64 == 0) {
    long long tiles = (long long)(M / 32) * (N / 64);
    k_gemm<4><<<(int)((tiles + 7) / 8), 256, 0, s>>>(A, lda, Wsw, N, kb, bias, Cf, Ch, ldc, M, relu);
  } else {
    long long tiles = (long long)(M / 32) * (N / 32);
    k_gemm<2><<<(int)((tiles + 7) / 8), 256, 0, s>>>(A, lda, Wsw, N, kb, bias, Cf, Ch, ldc, M, relu);
  }
}

struct MetaW {
  const h16 *We1, *We2, *WMkT, *WMv, *Wn1, *Wn2, *Wg1, *Wg2;
  const float *be1, *be2, *lng, *lnb, *bn1, *bn2, *bg1, *bg2;
};

extern "C" void kernel_launch(void* const* d_in, const int* in_sizes, int n_in,
                              void* d_out, int out_size, void* d_ws, size_t ws_size,
                              hipStream_t stream)
{
  (void)in_sizes; (void)n_in; (void)out_size; (void)ws_size;
  const float* sv_x  = (const float*)d_in[0];
  const int*   sv_ei = (const int*)  d_in[1];
  const float* sv_ea = (const float*)d_in[2];
  const int*   sv_bt = (const int*)  d_in[3];
  const float* sv_u  = (const float*)d_in[4];
  const float* sv_hb = (const float*)d_in[5];
  const float* so_x  = (const float*)d_in[6];
  const int*   so_ei = (const int*)  d_in[7];
  const float* so_ea = (const float*)d_in[8];
  const int*   so_bt = (const int*)  d_in[9];
  const float* so_u  = (const float*)d_in[10];
  const float* so_hb = (const float*)d_in[11];
  const float* Tarr  = (const float*)d_in[12];
  const float* const* pp = (const float* const*)(d_in + 13);
  // pp: 0 g1_edge_w1,1 b1,2 w2,3 b2 | 4 g1_ln_g,5 ln_b | 6 Mk,7 Mv | 8..11 g1_node
  //     12..15 g1_glob | 16..19 g2_edge | 20,21 g2_ln | 22,23 g2_Mk/Mv | 24..27 g2_node
  //     28..31 g2_glob | 32..34 gn1 w/b/ms | 35..37 gn2 | 38,39 mp_proj | 40,41 mp_e_w1/b1
  //     42,43 mp_e_w2/b2 | 44,45 mp_root | 46,47 gru_wih/bih | 48,49 gru_whh/bhh
  //     50..55 head a | 56..61 head b

  char* base = (char*)d_ws;
  size_t off = 0;
  auto alloc = [&](size_t bytes) -> void* {
    void* p = base + off; off = (off + bytes + 255) & ~(size_t)255; return p;
  };
  auto wcvt = [&](const float* src, h16* dst, int K0, int N0, int Kp, int Np, int mode) {
    int n = Kp * Np;
    k_wswz<<<(n + 255) / 256, 256, 0, stream>>>(src, dst, K0, N0, Kp, Np, mode);
  };

  // ---- persistent: swizzled f16 weights + per-branch results ----
  h16* W_e1  = (h16*)alloc(160 * 64 * 2);
  h16* W_e2  = (h16*)alloc(64 * 64 * 2);
  h16* W_Mk  = (h16*)alloc(160 * 128 * 2);
  h16* W_Mv  = (h16*)alloc(128 * 160 * 2);
  h16* W_n1  = (h16*)alloc(160 * 64 * 2);
  h16* W_n2  = (h16*)alloc(64 * 64 * 2);
  h16* W_gw1 = (h16*)alloc(160 * 64 * 2);
  h16* W_gw2 = (h16*)alloc(64 * 64 * 2);
  h16* X_e1  = (h16*)alloc(256 * 64 * 2);
  h16* X_e2  = (h16*)alloc(64 * 64 * 2);
  h16* X_Mk  = (h16*)alloc(192 * 128 * 2);
  h16* X_Mv  = (h16*)alloc(128 * 192 * 2);
  h16* X_n1  = (h16*)alloc(192 * 64 * 2);
  h16* X_n2  = (h16*)alloc(64 * 64 * 2);
  h16* X_gw1 = (h16*)alloc(192 * 64 * 2);
  h16* X_gw2 = (h16*)alloc(64 * 64 * 2);
  h16* Wproj = (h16*)alloc(128 * 128 * 2);
  h16* Wb2   = (h16*)alloc(128 * 128 * 2);
  h16* W2p   = (h16*)alloc((size_t)128 * 4096 * 2);
  h16* Wroot = (h16*)alloc(128 * 128 * 2);
  h16* WihT  = (h16*)alloc(128 * 384 * 2);
  h16* WhhT  = (h16*)alloc(128 * 384 * 2);
  h16* Wm1a  = (h16*)alloc(256 * 64 * 2);
  h16* Wm2a  = (h16*)alloc(64 * 64 * 2);
  h16* Wm1b  = (h16*)alloc(256 * 64 * 2);
  h16* Wm2b  = (h16*)alloc(64 * 64 * 2);
  float* xg1 = (float*)alloc((size_t)NB * 64 * 4);
  float* u1f = (float*)alloc((size_t)NB * 64 * 4);
  float* xg2 = (float*)alloc((size_t)NB * 64 * 4);
  float* u2f = (float*)alloc((size_t)NB * 64 * 4);

  wcvt(pp[0],  W_e1, 139, 64, 160, 64, 0);
  wcvt(pp[2],  W_e2, 64, 64, 64, 64, 0);
  wcvt(pp[6],  W_Mk, 131, 128, 160, 128, 1);   // Mk[128,131] -> Mk^T padded
  wcvt(pp[7],  W_Mv, 128, 131, 128, 160, 0);   // Mv padded cols
  wcvt(pp[8],  W_n1, 131, 64, 160, 64, 0);
  wcvt(pp[10], W_n2, 64, 64, 64, 64, 0);
  wcvt(pp[12], W_gw1, 131, 64, 160, 64, 0);
  wcvt(pp[14], W_gw2, 64, 64, 64, 64, 0);
  wcvt(pp[16], X_e1, 256, 64, 256, 64, 0);
  wcvt(pp[18], X_e2, 64, 64, 64, 64, 0);
  wcvt(pp[22], X_Mk, 192, 128, 192, 128, 1);
  wcvt(pp[23], X_Mv, 128, 192, 128, 192, 0);
  wcvt(pp[24], X_n1, 192, 64, 192, 64, 0);
  wcvt(pp[26], X_n2, 64, 64, 64, 64, 0);
  wcvt(pp[28], X_gw1, 192, 64, 192, 64, 0);
  wcvt(pp[30], X_gw2, 64, 64, 64, 64, 0);
  wcvt(pp[38], Wproj, 128, 128, 128, 128, 0);
  wcvt(pp[43], Wb2,   128, 128, 128, 128, 0);  // b2 viewed as [128,128]
  wcvt(pp[42], W2p,   128, 4096, 128, 4096, 2);// W2' reorder of mp_e_w2
  wcvt(pp[44], Wroot, 128, 128, 128, 128, 0);
  wcvt(pp[46], WihT, 128, 384, 128, 384, 1);   // gru_wih[384,128] -> transpose
  wcvt(pp[48], WhhT, 128, 384, 128, 384, 1);
  wcvt(pp[50], Wm1a, 256, 64, 256, 64, 0);
  wcvt(pp[52], Wm2a, 64, 64, 64, 64, 0);
  wcvt(pp[56], Wm1b, 256, 64, 256, 64, 0);
  wcvt(pp[58], Wm2b, 64, 64, 64, 64, 0);

  // ---- branch scratch arena (reused by system phase afterwards) ----
  size_t arena0 = off;
  h16*   Ein   = (h16*)  alloc((size_t)NEDGES * 256 * 2);          // 104.9 MB
  float* attn  = (float*)Ein;                                      // overlay (<= 78.7 MB)
  h16*   nh1   = (h16*)((char*)Ein + (size_t)NNODES * 192 * 4);    // overlay (+13.2 MB)
  h16*   Eh1   = (h16*)  alloc((size_t)NEDGES * 64 * 2);
  h16*   probs = Eh1;                                              // overlay, same size
  float* eaNew = (float*)alloc((size_t)NEDGES * 64 * 4);
  float* aggB  = (float*)alloc((size_t)NNODES * 64 * 4);
  float* outF  = (float*)alloc((size_t)NNODES * 192 * 4);
  h16*   outH  = (h16*)  alloc((size_t)NNODES * 192 * 2);
  float* logit = (float*)alloc((size_t)NNODES * 128 * 4);
  float* xNew  = (float*)alloc((size_t)NNODES * 64 * 4);
  float* nagg  = (float*)alloc((size_t)NB * 64 * 4);
  float* eagg  = (float*)alloc((size_t)NB * 64 * 4);
  h16*   Gin   = (h16*)  alloc((size_t)NB * 192 * 2);
  h16*   Gh1   = (h16*)  alloc((size_t)NB * 64 * 2);
  float* uTmp  = (float*)alloc((size_t)NB * 64 * 4);

  MetaW M1 = { W_e1, W_e2, W_Mk, W_Mv, W_n1, W_n2, W_gw1, W_gw2,
               pp[1], pp[3], pp[4], pp[5], pp[9], pp[11], pp[13], pp[15] };
  MetaW M2 = { X_e1, X_e2, X_Mk, X_Mv, X_n1, X_n2, X_gw1, X_gw2,
               pp[17], pp[19], pp[20], pp[21], pp[25], pp[27], pp[29], pp[31] };

  auto run_meta = [&](const float* x, const float* ea, int eaDim,
                      const float* u, int uDim, const int* row, const int* col,
                      const int* batch, const MetaW& Wm, int d1, int LD, int KE,
                      float* uOut)
  {
    { long long n = (long long)NEDGES * KE;
      k_edge_in<<<(int)((n + 255) / 256), 256, 0, stream>>>(x, ea, eaDim, u, uDim,
                                                            row, col, batch, Ein, KE); }
    gemm(stream, Ein, KE, Wm.We1, 64, Wm.be1, nullptr, Eh1, 64, NEDGES, KE, 1);
    gemm(stream, Eh1, 64, Wm.We2, 64, Wm.be2, eaNew, nullptr, 64, NEDGES, 64, 0);
    { long long n = (long long)NNODES * 64;
      k_zero<<<(int)((n + 255) / 256), 256, 0, stream>>>(aggB, n); }
    k_scatter<<<(NEDGES * 64 + 255) / 256, 256, 0, stream>>>(eaNew, col, aggB);
    k_node_ln<<<NNODES / 8, 256, 0, stream>>>(x, aggB, u, uDim, batch,
                                              Wm.lng, Wm.lnb, outF, outH, d1, LD);
    gemm(stream, outH, LD, Wm.WMkT, 128, nullptr, logit, nullptr, 128, NNODES, LD, 0);
    k_softmax128<<<NNODES / 8, 256, 0, stream>>>(logit, probs);
    gemm(stream, probs, 128, Wm.WMv, LD, nullptr, attn, nullptr, LD, NNODES, 128, 0);
    { long long n = (long long)NNODES * LD;
      k_mix<<<(int)((n + 255) / 256), 256, 0, stream>>>(attn, outF, outH, n); }
    gemm(stream, outH, LD, Wm.Wn1, 64, Wm.bn1, nullptr, nh1, 64, NNODES, LD, 1);
    gemm(stream, nh1, 64, Wm.Wn2, 64, Wm.bn2, xNew, nullptr, 64, NNODES, 64, 0);
    k_seg_mean<<<NB, 64, 0, stream>>>(xNew, nagg, NPG);
    k_seg_mean<<<NB, 64, 0, stream>>>(eaNew, eagg, EPG);
    k_glob_in<<<(NB * LD + 255) / 256, 256, 0, stream>>>(u, uDim, nagg, eagg, Gin, LD);
    gemm(stream, Gin, LD, Wm.Wg1, 64, Wm.bg1, nullptr, Gh1, 64, NB, LD, 1);
    gemm(stream, Gh1, 64, Wm.Wg2, 64, Wm.bg2, uOut, nullptr, 64, NB, 64, 0);
  };

  auto run_branch = [&](const float* x0, const int* ei, const float* ea0,
                        const int* batch, const float* u0, float* xgOut, float* uOut)
  {
    const int* row = ei;
    const int* col = ei + NEDGES;
    run_meta(x0, ea0, 12, u0, 3, row, col, batch, M1, 131, 160, 160, uTmp);
    k_gnorm<<<NB, 64, 0, stream>>>(xNew, pp[32], pp[33], pp[34]);
    run_meta(xNew, eaNew, 64, uTmp, 64, row, col, batch, M2, 192, 192, 256, uOut);
    k_gnorm<<<NB, 64, 0, stream>>>(xNew, pp[35], pp[36], pp[37]);
    k_seg_mean<<<NB, 64, 0, stream>>>(xNew, xgOut, NPG);
  };

  run_branch(sv_x, sv_ei, sv_ea, sv_bt, sv_u, xg1, u1f);
  run_branch(so_x, so_ei, so_ea, so_bt, so_u, xg2, u2f);

  // ---- system graph (reuse branch arena) ----
  off = arena0;
  h16*   nf16 = (h16*)  alloc((size_t)2 * NB * 128 * 2);
  float* hF   = (float*)alloc((size_t)2 * NB * 128 * 4);
  h16*   hH   = (h16*)  alloc((size_t)2 * NB * 128 * 2);
  float* ewh  = (float*)alloc((size_t)4 * NB * 32 * 4);
  float* Pm   = (float*)alloc((size_t)4096 * 4096 * 4);
  float* Qm   = (float*)alloc((size_t)2 * NB * 128 * 4);
  float* conv = (float*)alloc((size_t)2 * NB * 128 * 4);
  h16*   xt16 = (h16*)  alloc((size_t)2 * NB * 128 * 2);
  float* giB  = (float*)alloc((size_t)2 * NB * 384 * 4);
  float* ghB  = (float*)alloc((size_t)2 * NB * 384 * 4);
  h16*   hxg  = (h16*)  alloc((size_t)NB * 256 * 2);
  h16*   A1   = (h16*)  alloc((size_t)NB * 64 * 2);
  float* A2a  = (float*)alloc((size_t)NB * 64 * 4);
  float* A2b  = (float*)alloc((size_t)NB * 64 * 4);

  k_nodefeat<<<(2 * NB * 128 + 255) / 256, 256, 0, stream>>>(xg1, u1f, xg2, u2f, nf16);
  gemm(stream, nf16, 128, Wproj, 128, pp[39], hF, hH, 128, 2 * NB, 128, 1);
  k_ewh<<<(4 * NB * 32 + 255) / 256, 256, 0, stream>>>(sv_hb, so_hb, pp[40], pp[41], ewh);
  gemm(stream, hH, 128, W2p, 4096, nullptr, Pm, nullptr, 4096, 2 * NB, 128, 0);
  gemm(stream, hH, 128, Wb2, 128, nullptr, Qm, nullptr, 128, 2 * NB, 128, 0);
  gemm(stream, hH, 128, Wroot, 128, pp[45], conv, nullptr, 128, 2 * NB, 128, 0);
  k_convmsg<<<2 * NB, 128, 0, stream>>>(conv, Qm, Pm, ewh, xt16);
  gemm(stream, xt16, 128, WihT, 384, pp[47], giB, nullptr, 384, 2 * NB, 128, 0);
  gemm(stream, hH, 128, WhhT, 384, pp[49], ghB, nullptr, 384, 2 * NB, 128, 0);
  k_gru<<<(2 * NB * 128 + 255) / 256, 256, 0, stream>>>(giB, ghB, hF, hxg);
  gemm(stream, hxg, 256, Wm1a, 64, pp[51], nullptr, A1, 64, NB, 256, 1);
  gemm(stream, A1, 64, Wm2a, 64, pp[53], A2a, nullptr, 64, NB, 64, 1);
  gemm(stream, hxg, 256, Wm1b, 64, pp[57], nullptr, A1, 64, NB, 256, 1);
  gemm(stream, A1, 64, Wm2b, 64, pp[59], A2b, nullptr, 64, NB, 64, 1);
  k_head<<<(NB + 255) / 256, 256, 0, stream>>>(A2a, A2b, pp[54], pp[55], pp[60], pp[61],
                                               Tarr, (float*)d_out);
}
